// ETNNForecastingModel_85023172592610
// MI455X (gfx1250) — compile-verified, matrix operand order checked by reading, not run on hardware
//
#include <hip/hip_runtime.h>

// ---------------- problem constants (fixed by the reference) ----------------
constexpr int B   = 2;
constexpr int N   = 10000;
constexpr int E   = 100000;
constexpr int H   = 128;
constexpr int DIN = 8;
constexpr int DB  = 16;
constexpr int L   = 2;
constexpr int NB  = 3;
constexpr int K1  = 288;   // 2H+1 = 257 padded up to a multiple of 32

typedef __attribute__((ext_vector_type(16))) __bf16 v16bf;
typedef __attribute__((ext_vector_type(4)))  __bf16 bf16x4;
typedef __attribute__((ext_vector_type(2)))  __bf16 bf16x2;
typedef __attribute__((ext_vector_type(8)))  float  v8f;
typedef __attribute__((ext_vector_type(4)))  float  f32x4;

__device__ __forceinline__ float silu_f(float x) {
  return x * (1.0f / (1.0f + __expf(-x)));
}

__device__ __forceinline__ bf16x4 cvt_bf4(f32x4 v) {
  bf16x4 o;
  o.x = (__bf16)v.x; o.y = (__bf16)v.y; o.z = (__bf16)v.z; o.w = (__bf16)v.w;
  return o;
}

// ---- A fragment (16x32 bf16) from LDS, per ISA 16-bit A layout -------------
// lane 0..15  : M = lane, K = {0..7} ∪ {16..23}  (2 per VGPR)
// lane 16..31 : M = lane-16, K = {8..15} ∪ {24..31}
__device__ __forceinline__ v16bf load_a_frag_lds(const __bf16* Alds, int strideA,
                                                 int kt, int lane) {
  v16bf a;
  const int m     = lane & 15;
  const int khalf = (lane >> 4) * 8;
  const __bf16* rowp = Alds + m * strideA + kt * 32 + khalf;
#pragma unroll
  for (int q = 0; q < 8; ++q) {
    const int kb = (q < 4) ? (2 * q) : (16 + 2 * (q - 4));
    bf16x2 pr = *reinterpret_cast<const bf16x2*>(rowp + kb);  // merges to ds_load_b128
    a[2 * q]     = pr.x;
    a[2 * q + 1] = pr.y;
  }
  return a;
}

// ---- one 16x16 output tile: f32 acc of A[16xK] * W[Kx16] + bias ------------
// Wp points at pre-packed fragments: frag (kt, nt) = 32 lanes x 16 bf16 contig.
__device__ __forceinline__ v8f gemm_tile(const __bf16* Alds, int strideA, int KT,
                                         const __bf16* Wp, int nt, int lane,
                                         float bias) {
  v8f acc;
#pragma unroll
  for (int r = 0; r < 8; ++r) acc[r] = bias;
  const v16bf* fr = reinterpret_cast<const v16bf*>(Wp);
  for (int kt = 0; kt < KT; ++kt) {
    v16bf a = load_a_frag_lds(Alds, strideA, kt, lane);
    v16bf w = fr[(kt * 8 + nt) * 32 + lane];                 // global_load_b128 x2
    acc = __builtin_amdgcn_wmma_f32_16x16x32_bf16(false, a, false, w,
                                                  (short)0, acc, false, false);
  }
  return acc;
}

// ---------------- weight pre-pack: f32 [K x Ncols] -> B-fragment bf16 -------
__global__ void pack_w_kernel(const float* __restrict__ src, __bf16* __restrict__ dst,
                              int Ksrc, int total, int Ncols) {
  int idx = blockIdx.x * 256 + threadIdx.x;
  if (idx >= total) return;
  const int hf   = idx & 1;
  const int q    = (idx >> 1) & 7;
  const int lane = (idx >> 4) & 31;
  const int rest = idx >> 9;
  const int NT   = Ncols >> 4;
  const int nt   = rest % NT;
  const int kt   = rest / NT;
  const int n    = nt * 16 + (lane & 15);
  const int khalf = (lane >> 4) * 8;
  const int kb   = (q < 4) ? (2 * q) : (16 + 2 * (q - 4));
  const int k    = kt * 32 + khalf + kb + hf;
  float val = (k < Ksrc) ? src[k * Ncols + n] : 0.0f;
  dst[idx] = (__bf16)val;
}

// ---------------- degree: count + reciprocal --------------------------------
__global__ void count_kernel(const int* __restrict__ edst_all, float* __restrict__ cnt) {
  int idx = blockIdx.x * 256 + threadIdx.x;
  if (idx >= NB * E) return;
  int k = idx / E;
  atomicAdd(&cnt[k * N + edst_all[idx]], 1.0f);
}

__global__ void recip_kernel(const float* __restrict__ cnt, float* __restrict__ dinv) {
  int idx = blockIdx.x * 256 + threadIdx.x;
  if (idx >= NB * N) return;
  dinv[idx] = 1.0f / fmaxf(cnt[idx], 1.0f);
}

// ---------------- h / p initialization --------------------------------------
__global__ void init_h_kernel(const float* __restrict__ x, const float* __restrict__ basef,
                              const float* __restrict__ Win, const float* __restrict__ b_in,
                              const int* __restrict__ eidx, float* __restrict__ h) {
  int idx = blockIdx.x * 256 + threadIdx.x;
  if (idx >= B * N * H) return;
  const int c   = idx & (H - 1);
  const int row = idx >> 7;
  const int b   = row / N;
  const int n   = row - b * N;
  float acc = b_in[c];
#pragma unroll
  for (int j = 0; j < DB; ++j) acc += basef[n * DB + j] * Win[j * H + c];
  if (n == eidx[b]) {
#pragma unroll
    for (int j = 0; j < DIN; ++j) acc += x[b * DIN + j] * Win[(DB + j) * H + c];
  }
  h[idx] = acc;
}

__global__ void init_p_kernel(const float* __restrict__ basep, float* __restrict__ p0) {
  int idx = blockIdx.x * 256 + threadIdx.x;
  if (idx >= B * N * 3) return;
  p0[idx] = basep[idx % (N * 3)];
}

// ---------------- fused per-edge message/position kernel --------------------
// grid = (E/16, B), block = 256 (8 waves). One workgroup owns 16 edges.
__global__ __launch_bounds__(256)
void edge_msg_kernel(const float* __restrict__ h, const float* __restrict__ p,
                     float* __restrict__ p_nxt, float* __restrict__ m_sum,
                     const float* __restrict__ deginv,
                     const int* __restrict__ esrc, const int* __restrict__ edst,
                     const __bf16* __restrict__ W1p, const float* __restrict__ b1,
                     const __bf16* __restrict__ W2p, const float* __restrict__ b2,
                     const __bf16* __restrict__ PW1p, const float* __restrict__ pb1,
                     const float* __restrict__ pW2, const float* __restrict__ pb2) {
  __shared__ __align__(16) __bf16 feat[16 * K1];
  __shared__ __align__(16) __bf16 m1buf[16 * H];
  __shared__ __align__(16) __bf16 mbuf[16 * H];
  __shared__ __align__(16) float  t1buf[16 * H];
  __shared__ int   dstIdx[16];
  __shared__ int   srcIdx[16];
  __shared__ float dinvS[16];
  __shared__ float d2S[16];
  __shared__ float relS[16][3];
  __shared__ float coefS[16];

  const int tid  = threadIdx.x;
  const int lane = tid & 31;
  const int wave = tid >> 5;
  const int b    = blockIdx.y;
  const int e0   = blockIdx.x * 16;

  // warm GL2/WGP$ with this tile's weights while we gather features
  __builtin_prefetch(W1p + (size_t)tid * 64, 0, 0);    // global_prefetch_b8
  __builtin_prefetch(W2p + (size_t)tid * 64, 0, 0);
  __builtin_prefetch(PW1p + (size_t)tid * 64, 0, 0);

  if (tid < 16) {
    const int e = e0 + tid;
    const int d = edst[e];
    const int s = esrc[e];
    dstIdx[tid] = d;
    srcIdx[tid] = s;
    dinvS[tid]  = deginv[d];
    float d2 = 0.0f;
#pragma unroll
    for (int c = 0; c < 3; ++c) {
      float r = p[((size_t)b * N + d) * 3 + c] - p[((size_t)b * N + s) * 3 + c];
      relS[tid][c] = r;
      d2 += r * r;
    }
    d2S[tid] = d2;
  }
  __syncthreads();

  // gather [h_dst | h_src | d2 | 0-pad] as bf16 into LDS, 4 elements at a time
  {
    const f32x4* hv = reinterpret_cast<const f32x4*>(h);
    constexpr int CC = K1 / 4;           // 72 chunks per row
    for (int idx = tid; idx < 16 * CC; idx += 256) {
      const int row = idx / CC;
      const int cc  = idx - row * CC;
      f32x4 v;
      if (cc < H / 4) {
        v = hv[((size_t)b * N + dstIdx[row]) * (H / 4) + cc];
      } else if (cc < 2 * H / 4) {
        v = hv[((size_t)b * N + srcIdx[row]) * (H / 4) + (cc - H / 4)];
      } else if (cc == 2 * H / 4) {
        v.x = d2S[row]; v.y = 0.0f; v.z = 0.0f; v.w = 0.0f;
      } else {
        v.x = 0.0f; v.y = 0.0f; v.z = 0.0f; v.w = 0.0f;
      }
      *reinterpret_cast<bf16x4*>(&feat[row * K1 + cc * 4]) = cvt_bf4(v);
    }
  }
  __syncthreads();

  const int Mbase = (lane >> 4) * 8;
  const int col   = wave * 16 + (lane & 15);

  // GEMM1: m1 = silu(feat @ W1 + b1)   (K = 288)
  v8f acc = gemm_tile(feat, K1, K1 / 32, W1p, wave, lane, b1[col]);
#pragma unroll
  for (int r = 0; r < 8; ++r)
    m1buf[(Mbase + r) * H + col] = (__bf16)silu_f(acc[r]);
  __syncthreads();

  // GEMM2: m = silu(m1 @ W2 + b2)      (K = 128), scatter m/deg into m_sum
  acc = gemm_tile(m1buf, H, H / 32, W2p, wave, lane, b2[col]);
#pragma unroll
  for (int r = 0; r < 8; ++r) {
    const int M = Mbase + r;
    float val = silu_f(acc[r]);
    mbuf[M * H + col] = (__bf16)val;
    atomicAdd(&m_sum[((size_t)b * N + dstIdx[M]) * H + col], val * dinvS[M]);
  }
  __syncthreads();

  // GEMM3: t1 = silu(m @ pos_W1 + pb1) (K = 128)
  acc = gemm_tile(mbuf, H, H / 32, PW1p, wave, lane, pb1[col]);
#pragma unroll
  for (int r = 0; r < 8; ++r)
    t1buf[(Mbase + r) * H + col] = silu_f(acc[r]);
  __syncthreads();

  // coef = t1 @ pos_W2 + pb2   (128 -> 1 per edge)
  if (tid < 16) {
    float a0 = pb2[0];
    for (int c = 0; c < H; ++c) a0 += t1buf[tid * H + c] * pW2[c];
    coefS[tid] = a0;
  }
  __syncthreads();

  // p_new[dst] += rel * coef / deg
  if (tid < 48) {
    const int r = tid / 3;
    const int c = tid - r * 3;
    atomicAdd(&p_nxt[((size_t)b * N + dstIdx[r]) * 3 + c],
              relS[r][c] * coefS[r] * dinvS[r]);
  }
}

// ---------------- node update: h += MLP([h | m_sum]) ------------------------
// grid = B*N/16, block = 256. Rows exclusive -> no atomics.
__global__ __launch_bounds__(256)
void node_update_kernel(float* __restrict__ h, const float* __restrict__ m_sum,
                        const __bf16* __restrict__ UW1p, const float* __restrict__ ub1,
                        const __bf16* __restrict__ UW2p, const float* __restrict__ ub2) {
  __shared__ __align__(16) __bf16 feat[16 * 256];
  __shared__ __align__(16) __bf16 tbuf[16 * H];

  const int tid  = threadIdx.x;
  const int lane = tid & 31;
  const int wave = tid >> 5;
  const size_t row0 = (size_t)blockIdx.x * 16;

  {
    const f32x4* hv = reinterpret_cast<const f32x4*>(h);
    const f32x4* mv = reinterpret_cast<const f32x4*>(m_sum);
    for (int idx = tid; idx < 16 * 64; idx += 256) {
      const int r  = idx >> 6;
      const int cc = idx & 63;
      f32x4 v = (cc < 32) ? hv[(row0 + r) * (H / 4) + cc]
                          : mv[(row0 + r) * (H / 4) + (cc - 32)];
      *reinterpret_cast<bf16x4*>(&feat[r * 256 + cc * 4]) = cvt_bf4(v);
    }
  }
  __syncthreads();

  const int Mbase = (lane >> 4) * 8;
  const int col   = wave * 16 + (lane & 15);

  v8f acc = gemm_tile(feat, 256, 8, UW1p, wave, lane, ub1[col]);
#pragma unroll
  for (int r = 0; r < 8; ++r)
    tbuf[(Mbase + r) * H + col] = (__bf16)silu_f(acc[r]);
  __syncthreads();

  acc = gemm_tile(tbuf, H, 4, UW2p, wave, lane, ub2[col]);
#pragma unroll
  for (int r = 0; r < 8; ++r)
    h[(row0 + Mbase + r) * H + col] += acc[r];   // residual add (no silu)
}

// ---------------- readout ---------------------------------------------------
__global__ void readout_kernel(const float* __restrict__ h, const int* __restrict__ eidx,
                               const float* __restrict__ W1, const float* __restrict__ ob1,
                               const float* __restrict__ W2, const float* __restrict__ ob2,
                               float* __restrict__ out) {
  __shared__ float ev[H];
  __shared__ float tv[H];
  const int b   = blockIdx.x;
  const int tid = threadIdx.x;    // H threads
  const int n   = eidx[b];
  ev[tid] = h[((size_t)b * N + n) * H + tid];
  __syncthreads();
  float a = ob1[tid];
  for (int j = 0; j < H; ++j) a += ev[j] * W1[j * H + tid];
  tv[tid] = silu_f(a);
  __syncthreads();
  if (tid == 0) {
    float o = ob2[0];
    for (int c = 0; c < H; ++c) o += tv[c] * W2[c];
    out[b] = o;
  }
}

// ---------------- host orchestration ----------------------------------------
extern "C" void kernel_launch(void* const* d_in, const int* in_sizes, int n_in,
                              void* d_out, int out_size, void* d_ws, size_t ws_size,
                              hipStream_t stream) {
  (void)in_sizes; (void)n_in; (void)out_size; (void)ws_size;

  const float* x     = (const float*)d_in[0];
  const float* basef = (const float*)d_in[1];
  const float* basep = (const float*)d_in[2];
  const float* Win   = (const float*)d_in[3];
  const float* b_in  = (const float*)d_in[4];
  const float* msgW1 = (const float*)d_in[5];
  const float* msgb1 = (const float*)d_in[6];
  const float* msgW2 = (const float*)d_in[7];
  const float* msgb2 = (const float*)d_in[8];
  const float* posW1 = (const float*)d_in[9];
  const float* posb1 = (const float*)d_in[10];
  const float* posW2 = (const float*)d_in[11];
  const float* posb2 = (const float*)d_in[12];
  const float* updW1 = (const float*)d_in[13];
  const float* updb1 = (const float*)d_in[14];
  const float* updW2 = (const float*)d_in[15];
  const float* updb2 = (const float*)d_in[16];
  const float* outW1 = (const float*)d_in[17];
  const float* outb1 = (const float*)d_in[18];
  const float* outW2 = (const float*)d_in[19];
  const float* outb2 = (const float*)d_in[20];
  const int*   eidx  = (const int*)d_in[21];
  const int*   esrc  = (const int*)d_in[22];
  const int*   edst  = (const int*)d_in[23];

  char* ws = (char*)d_ws;
  size_t off = 0;
  auto carve = [&](size_t bytes) -> char* {
    char* ptr = ws + off;
    off = (off + bytes + 255) & ~(size_t)255;
    return ptr;
  };

  float*  h     = (float*)carve((size_t)B * N * H * 4);
  float*  m_sum = (float*)carve((size_t)B * N * H * 4);
  float*  p0    = (float*)carve((size_t)B * N * 3 * 4);
  float*  p1    = (float*)carve((size_t)B * N * 3 * 4);
  float*  cnt   = (float*)carve((size_t)NB * N * 4);
  float*  dinv  = (float*)carve((size_t)NB * N * 4);
  __bf16* W1p   = (__bf16*)carve((size_t)L * NB * K1 * H * 2);
  __bf16* W2p   = (__bf16*)carve((size_t)L * NB * H * H * 2);
  __bf16* PW1p  = (__bf16*)carve((size_t)L * NB * H * H * 2);
  __bf16* UW1p  = (__bf16*)carve((size_t)L * 2 * H * H * 2);
  __bf16* UW2p  = (__bf16*)carve((size_t)L * H * H * 2);

  // --- pack all weights into WMMA B-fragment order (bf16) ---
  for (int l = 0; l < L; ++l) {
    for (int k = 0; k < NB; ++k) {
      const int lk = l * NB + k;
      pack_w_kernel<<<(K1 * H + 255) / 256, 256, 0, stream>>>(
          msgW1 + (size_t)lk * (2 * H + 1) * H, W1p + (size_t)lk * K1 * H,
          2 * H + 1, K1 * H, H);
      pack_w_kernel<<<(H * H + 255) / 256, 256, 0, stream>>>(
          msgW2 + (size_t)lk * H * H, W2p + (size_t)lk * H * H, H, H * H, H);
      pack_w_kernel<<<(H * H + 255) / 256, 256, 0, stream>>>(
          posW1 + (size_t)lk * H * H, PW1p + (size_t)lk * H * H, H, H * H, H);
    }
    pack_w_kernel<<<(2 * H * H + 255) / 256, 256, 0, stream>>>(
        updW1 + (size_t)l * 2 * H * H, UW1p + (size_t)l * 2 * H * H,
        2 * H, 2 * H * H, H);
    pack_w_kernel<<<(H * H + 255) / 256, 256, 0, stream>>>(
        updW2 + (size_t)l * H * H, UW2p + (size_t)l * H * H, H, H * H, H);
  }

  // --- degree reciprocals (same across layers/batches) ---
  hipMemsetAsync(cnt, 0, (size_t)NB * N * 4, stream);
  count_kernel<<<(NB * E + 255) / 256, 256, 0, stream>>>(edst, cnt);
  recip_kernel<<<(NB * N + 255) / 256, 256, 0, stream>>>(cnt, dinv);

  // --- init h and p ---
  init_h_kernel<<<(B * N * H + 255) / 256, 256, 0, stream>>>(x, basef, Win, b_in, eidx, h);
  init_p_kernel<<<(B * N * 3 + 255) / 256, 256, 0, stream>>>(basep, p0);

  float* pc = p0;
  float* pn = p1;
  for (int l = 0; l < L; ++l) {
    hipMemsetAsync(m_sum, 0, (size_t)B * N * H * 4, stream);
    hipMemcpyAsync(pn, pc, (size_t)B * N * 3 * 4, hipMemcpyDeviceToDevice, stream);
    for (int k = 0; k < NB; ++k) {
      const int lk = l * NB + k;
      edge_msg_kernel<<<dim3(E / 16, B), 256, 0, stream>>>(
          h, pc, pn, m_sum, dinv + (size_t)k * N,
          esrc + (size_t)k * E, edst + (size_t)k * E,
          W1p + (size_t)lk * K1 * H, msgb1 + (size_t)lk * H,
          W2p + (size_t)lk * H * H, msgb2 + (size_t)lk * H,
          PW1p + (size_t)lk * H * H, posb1 + (size_t)lk * H,
          posW2 + (size_t)lk * H, posb2 + (size_t)lk);
    }
    node_update_kernel<<<(B * N) / 16, 256, 0, stream>>>(
        h, m_sum,
        UW1p + (size_t)l * 2 * H * H, updb1 + (size_t)l * H,
        UW2p + (size_t)l * H * H, updb2 + (size_t)l * H);
    float* t = pc; pc = pn; pn = t;
  }

  readout_kernel<<<B, H, 0, stream>>>(h, eidx, outW1, outb1, outW2, outb2, (float*)d_out);
}
